// SurfConvEncoder_40999757808031
// MI455X (gfx1250) — matrix operand order repaired and checked by reading, not compiled, not run on hardware
//
#include <hip/hip_runtime.h>
#include <hip/hip_bf16.h>
#include <math.h>

#define FEAT 128
#define HID  128
#define OUTD 64

typedef __attribute__((ext_vector_type(16))) __bf16 v16bf;
typedef __attribute__((ext_vector_type(8)))  float  v8f;
typedef __attribute__((ext_vector_type(4)))  float  v4f;

__device__ __forceinline__ v8f wmma_bf16(v16bf a, v16bf b, v8f c) {
  // D = A(16x32 bf16) * B(32x16 bf16) + C(16x16 f32)
  return __builtin_amdgcn_wmma_f32_16x16x32_bf16(false, a, false, b, (short)0, c,
                                                 false, false);
}

__device__ __forceinline__ v16bf cvt_frag(v4f f0, v4f f1, v4f f2, v4f f3) {
  v16bf a;
#pragma unroll
  for (int i = 0; i < 4; ++i) {
    a[i]      = (__bf16)f0[i];
    a[4 + i]  = (__bf16)f1[i];
    a[8 + i]  = (__bf16)f2[i];
    a[12 + i] = (__bf16)f3[i];
  }
  return a;
}

// Lane-dependent base pointer for a 16(M)x32(K) A-fragment (row-major f32 src).
// Lanes 0-15: M=0..15 / K kb+0..7 & kb+16..23 ; lanes 16-31: same rows, K +8.
__device__ __forceinline__ const float* a_ptr(const float* __restrict__ src,
                                              int mb, int n, int ld, int kt) {
  const int lane = threadIdx.x & 31;
  int m = mb + (lane & 15);
  if (m > n - 1) m = n - 1;  // clamp for partial M-tiles
  const int kb = (kt << 5) + ((lane >> 4) << 3);
  return src + (size_t)m * ld + kb;
}

__device__ __forceinline__ v16bf load_a(const float* p) {
  // two contiguous 8-float runs -> 4x global_load_b128
  return cvt_frag(*(const v4f*)(p + 0),  *(const v4f*)(p + 4),
                  *(const v4f*)(p + 16), *(const v4f*)(p + 20));
}

__device__ __forceinline__ v16bf load_a_nt(const float* p) {
  return cvt_frag(__builtin_nontemporal_load((const v4f*)(p + 0)),
                  __builtin_nontemporal_load((const v4f*)(p + 4)),
                  __builtin_nontemporal_load((const v4f*)(p + 16)),
                  __builtin_nontemporal_load((const v4f*)(p + 20)));
}

// A-fragment of zmix = 0.9*agg + 0.1*h0, computed on the fly (resid folded in).
__device__ __forceinline__ v16bf load_a_mix(const float* pz, const float* ph) {
  return cvt_frag(0.9f * (*(const v4f*)(pz + 0))  + 0.1f * (*(const v4f*)(ph + 0)),
                  0.9f * (*(const v4f*)(pz + 4))  + 0.1f * (*(const v4f*)(ph + 4)),
                  0.9f * (*(const v4f*)(pz + 16)) + 0.1f * (*(const v4f*)(ph + 16)),
                  0.9f * (*(const v4f*)(pz + 20)) + 0.1f * (*(const v4f*)(ph + 20)));
}

// ---------------- elementwise / graph kernels ----------------

__global__ __launch_bounds__(256) void init_deg_kernel(float* deg, int n) {
  int i = blockIdx.x * 256 + threadIdx.x;
  if (i < n) deg[i] = 1.0f;  // self-loop weight
}

__global__ __launch_bounds__(256) void deg_accum_kernel(float* __restrict__ deg,
    const int* __restrict__ col, const float* __restrict__ w, int E) {
  int e = blockIdx.x * 256 + threadIdx.x;
  if (e < E) atomicAdd(&deg[col[e]], w[e]);
}

__global__ __launch_bounds__(256) void dinv_kernel(float* deg, int n) {
  int i = blockIdx.x * 256 + threadIdx.x;
  if (i < n) {
    float d = deg[i];
    deg[i] = (d > 0.f) ? rsqrtf(d) : 0.f;
  }
}

// z[i,:] = dinv[i]^2 * h[i,:]   (self-loop term; also initializes z)
__global__ __launch_bounds__(256) void self_init_kernel(float* __restrict__ z,
    const float* __restrict__ h, const float* __restrict__ dinv, int n) {
  int i = blockIdx.x * 256 + threadIdx.x;
  if (i < n * HID) {
    float di = dinv[i >> 7];
    z[i] = di * di * h[i];
  }
}

// One wave per edge: gather h[row] (b128 per lane), scale, scatter-add into
// z[col] with f32 global atomics. h/z are L2-resident (51MB each vs 192MB L2).
__global__ __launch_bounds__(256) void edge_agg_kernel(float* __restrict__ z,
    const float* __restrict__ h, const float* __restrict__ dinv,
    const int* __restrict__ row, const int* __restrict__ col,
    const float* __restrict__ ea, int E) {
  int e = blockIdx.x * 8 + (threadIdx.x >> 5);
  if (e >= E) return;
  const int lane = threadIdx.x & 31;
  const int r = row[e], c = col[e];
  const float nrm = dinv[r] * ea[e] * dinv[c];
  const v4f v = ((const v4f*)(h + (size_t)r * HID))[lane];
  float* zc = z + (size_t)c * HID + lane * 4;
  atomicAdd(zc + 0, nrm * v[0]);
  atomicAdd(zc + 1, nrm * v[1]);
  atomicAdd(zc + 2, nrm * v[2]);
  atomicAdd(zc + 3, nrm * v[3]);
}

// ---------------- WMMA GEMM kernels ----------------
// Block = 256 threads = 8 waves; each wave owns a 16-row stripe; block covers
// 128 rows x full output width. Weights staged once per block into LDS,
// pre-swizzled into WMMA B-fragment layout: frag(nt,kt) -> lane%16 = N column, lane/16 = K +8 offset within
// the 32-K slab, 16 contiguous bf16 = K values (A's k order) => two ds_load_b128 per frag.

// h = relu(x @ W_in^T + b_in);  h0 = h
__global__ __launch_bounds__(256) void gemm_in_kernel(
    const float* __restrict__ x, const float* __restrict__ Win,
    const float* __restrict__ bin, float* __restrict__ h,
    float* __restrict__ h0, int n) {
  __shared__ __bf16 Bs[8 * 4 * 32 * 16];  // [nt][kt][lane][16] = 16384
  const int tid = threadIdx.x;
  for (int idx = tid; idx < 8 * 4 * 32 * 16; idx += 256) {
    const int e = idx & 15, l = (idx >> 4) & 31;
    const int kt = (idx >> 9) & 3, nt = idx >> 11;
    const int k = (kt << 5) + ((e >> 3) << 4) + ((l >> 4) << 3) + (e & 7);
    const int j = (nt << 4) + (l & 15);
    Bs[idx] = (__bf16)Win[j * FEAT + k];  // B[k][j] = Win[j][k]
  }
  __syncthreads();
  const int wave = tid >> 5, lane = tid & 31;
  const int mb = (blockIdx.x * 8 + wave) << 4;
  if (mb >= n) return;
  const v16bf* Bv = (const v16bf*)Bs;
  v16bf a0 = load_a_nt(a_ptr(x, mb, n, FEAT, 0));  // x is read-once: NT loads
  v16bf a1 = load_a_nt(a_ptr(x, mb, n, FEAT, 1));
  v16bf a2 = load_a_nt(a_ptr(x, mb, n, FEAT, 2));
  v16bf a3 = load_a_nt(a_ptr(x, mb, n, FEAT, 3));
  const int colLane = lane & 15;
  const int rhi = (lane >> 4) << 3;
  const bool full = (mb + 16 <= n);
#pragma unroll
  for (int nt = 0; nt < 8; ++nt) {
    const int col = (nt << 4) + colLane;
    const float bias = bin[col];
    v8f c;
#pragma unroll
    for (int r = 0; r < 8; ++r) c[r] = bias;
    c = wmma_bf16(a0, Bv[(nt * 4 + 0) * 32 + lane], c);
    c = wmma_bf16(a1, Bv[(nt * 4 + 1) * 32 + lane], c);
    c = wmma_bf16(a2, Bv[(nt * 4 + 2) * 32 + lane], c);
    c = wmma_bf16(a3, Bv[(nt * 4 + 3) * 32 + lane], c);
    if (full) {  // fast path: unguarded strided stores (781/782 blocks)
      float* ph  = h  + (size_t)(mb + rhi) * HID + col;
      float* ph0 = h0 + (size_t)(mb + rhi) * HID + col;
#pragma unroll
      for (int r = 0; r < 8; ++r) {
        const float v = c[r] > 0.f ? c[r] : 0.f;
        ph[(size_t)r * HID] = v;
        ph0[(size_t)r * HID] = v;
      }
    } else {
#pragma unroll
      for (int r = 0; r < 8; ++r) {
        const int rw = mb + rhi + r;
        if (rw < n) {
          const float v = c[r] > 0.f ? c[r] : 0.f;
          h[(size_t)rw * HID + col] = v;
          h0[(size_t)rw * HID + col] = v;
        }
      }
    }
  }
}

// h = relu((1-beta)*zmix + beta*(zmix @ W)), zmix = 0.9*agg + 0.1*h0 computed
// on the fly; beta folded into LDS copy of W, (1-beta)*zmix folded into C.
__global__ __launch_bounds__(256) void gemm_layer_kernel(
    const float* __restrict__ z, const float* __restrict__ h0,
    const float* __restrict__ W, float beta, float* __restrict__ hout, int n) {
  __shared__ __bf16 Bs[8 * 4 * 32 * 16];
  const int tid = threadIdx.x;
  for (int idx = tid; idx < 8 * 4 * 32 * 16; idx += 256) {
    const int e = idx & 15, l = (idx >> 4) & 31;
    const int kt = (idx >> 9) & 3, nt = idx >> 11;
    const int k = (kt << 5) + ((e >> 3) << 4) + ((l >> 4) << 3) + (e & 7);
    const int j = (nt << 4) + (l & 15);
    Bs[idx] = (__bf16)(beta * W[k * HID + j]);  // B[k][j] = beta * W[k][j]
  }
  __syncthreads();
  const int wave = tid >> 5, lane = tid & 31;
  const int mb = (blockIdx.x * 8 + wave) << 4;
  if (mb >= n) return;
  const v16bf* Bv = (const v16bf*)Bs;
  v16bf a0 = load_a_mix(a_ptr(z, mb, n, HID, 0), a_ptr(h0, mb, n, HID, 0));
  v16bf a1 = load_a_mix(a_ptr(z, mb, n, HID, 1), a_ptr(h0, mb, n, HID, 1));
  v16bf a2 = load_a_mix(a_ptr(z, mb, n, HID, 2), a_ptr(h0, mb, n, HID, 2));
  v16bf a3 = load_a_mix(a_ptr(z, mb, n, HID, 3), a_ptr(h0, mb, n, HID, 3));
  const float omb = 1.0f - beta;
  const int colLane = lane & 15;
  const int rhi = (lane >> 4) << 3;
  const bool full = (mb + 16 <= n);
#pragma unroll
  for (int nt = 0; nt < 8; ++nt) {
    const int col = (nt << 4) + colLane;
    v8f c;
#pragma unroll
    for (int r = 0; r < 8; ++r) {
      int rw = mb + rhi + r;
      if (rw > n - 1) rw = n - 1;
      const size_t ix = (size_t)rw * HID + col;
      c[r] = omb * (0.9f * z[ix] + 0.1f * h0[ix]);
    }
    c = wmma_bf16(a0, Bv[(nt * 4 + 0) * 32 + lane], c);
    c = wmma_bf16(a1, Bv[(nt * 4 + 1) * 32 + lane], c);
    c = wmma_bf16(a2, Bv[(nt * 4 + 2) * 32 + lane], c);
    c = wmma_bf16(a3, Bv[(nt * 4 + 3) * 32 + lane], c);
    if (full) {
      float* ph = hout + (size_t)(mb + rhi) * HID + col;
#pragma unroll
      for (int r = 0; r < 8; ++r) ph[(size_t)r * HID] = c[r] > 0.f ? c[r] : 0.f;
    } else {
#pragma unroll
      for (int r = 0; r < 8; ++r) {
        const int rw = mb + rhi + r;
        if (rw < n) hout[(size_t)rw * HID + col] = c[r] > 0.f ? c[r] : 0.f;
      }
    }
  }
}

// out = h @ W_out^T + b_out   (N x 64), write-once output -> NT stores
__global__ __launch_bounds__(256) void gemm_out_kernel(
    const float* __restrict__ h, const float* __restrict__ Wout,
    const float* __restrict__ bout, float* __restrict__ out, int n) {
  __shared__ __bf16 Bs[4 * 4 * 32 * 16];  // 8192
  const int tid = threadIdx.x;
  for (int idx = tid; idx < 4 * 4 * 32 * 16; idx += 256) {
    const int e = idx & 15, l = (idx >> 4) & 31;
    const int kt = (idx >> 9) & 3, nt = idx >> 11;
    const int k = (kt << 5) + ((e >> 3) << 4) + ((l >> 4) << 3) + (e & 7);
    const int j = (nt << 4) + (l & 15);   // 0..63
    Bs[idx] = (__bf16)Wout[j * HID + k];  // B[k][j] = Wout[j][k]
  }
  __syncthreads();
  const int wave = tid >> 5, lane = tid & 31;
  const int mb = (blockIdx.x * 8 + wave) << 4;
  if (mb >= n) return;
  const v16bf* Bv = (const v16bf*)Bs;
  v16bf a0 = load_a(a_ptr(h, mb, n, HID, 0));
  v16bf a1 = load_a(a_ptr(h, mb, n, HID, 1));
  v16bf a2 = load_a(a_ptr(h, mb, n, HID, 2));
  v16bf a3 = load_a(a_ptr(h, mb, n, HID, 3));
  const int colLane = lane & 15;
  const int rhi = (lane >> 4) << 3;
  const bool full = (mb + 16 <= n);
#pragma unroll
  for (int nt = 0; nt < 4; ++nt) {
    const int col = (nt << 4) + colLane;
    const float bias = bout[col];
    v8f c;
#pragma unroll
    for (int r = 0; r < 8; ++r) c[r] = bias;
    c = wmma_bf16(a0, Bv[(nt * 4 + 0) * 32 + lane], c);
    c = wmma_bf16(a1, Bv[(nt * 4 + 1) * 32 + lane], c);
    c = wmma_bf16(a2, Bv[(nt * 4 + 2) * 32 + lane], c);
    c = wmma_bf16(a3, Bv[(nt * 4 + 3) * 32 + lane], c);
    if (full) {
      float* po = out + (size_t)(mb + rhi) * OUTD + col;
#pragma unroll
      for (int r = 0; r < 8; ++r)
        __builtin_nontemporal_store(c[r], po + (size_t)r * OUTD);
    } else {
#pragma unroll
      for (int r = 0; r < 8; ++r) {
        const int rw = mb + rhi + r;
        if (rw < n)
          __builtin_nontemporal_store(c[r], out + (size_t)rw * OUTD + col);
      }
    }
  }
}

// ---------------- driver ----------------

extern "C" void kernel_launch(void* const* d_in, const int* in_sizes, int n_in,
                              void* d_out, int out_size, void* d_ws, size_t ws_size,
                              hipStream_t stream) {
  const float* x    = (const float*)d_in[0];
  const int*   ei   = (const int*)d_in[1];
  const float* ea   = (const float*)d_in[2];
  const float* Win  = (const float*)d_in[3];
  const float* bin  = (const float*)d_in[4];
  const float* W1   = (const float*)d_in[5];
  const float* W2   = (const float*)d_in[6];
  const float* W3   = (const float*)d_in[7];
  const float* Wout = (const float*)d_in[8];
  const float* bout = (const float*)d_in[9];
  float* out = (float*)d_out;

  const int n = in_sizes[0] / FEAT;   // 100000
  const int E = in_sizes[2];          // 1600000
  const int* rowi = ei;               // edge_index[0]
  const int* coli = ei + E;           // edge_index[1]

  // Workspace layout (floats): dinv[nAl] | h[nAl*128] | h0[nAl*128] | z[nAl*128]
  float* ws = (float*)d_ws;
  const size_t nAl = ((size_t)n + 127) & ~(size_t)127;
  float* dinv = ws;
  float* h    = ws + nAl;
  float* h0   = h + nAl * HID;
  float* z    = h0 + nAl * HID;

  dim3 blk(256);
  init_deg_kernel<<<(n + 255) / 256, blk, 0, stream>>>(dinv, n);
  deg_accum_kernel<<<(E + 255) / 256, blk, 0, stream>>>(dinv, coli, ea, E);
  dinv_kernel<<<(n + 255) / 256, blk, 0, stream>>>(dinv, n);

  gemm_in_kernel<<<(n + 127) / 128, blk, 0, stream>>>(x, Win, bin, h, h0, n);

  const float* Ws[3] = {W1, W2, W3};
  for (int l = 0; l < 3; ++l) {
    const float beta = logf(0.5f / (float)(l + 1) + 1.0f);
    self_init_kernel<<<(n * HID + 255) / 256, blk, 0, stream>>>(z, h, dinv, n);
    edge_agg_kernel<<<(E + 7) / 8, blk, 0, stream>>>(z, h, dinv, rowi, coli, ea, E);
    gemm_layer_kernel<<<(n + 127) / 128, blk, 0, stream>>>(z, h0, Ws[l], beta, h, n);
  }

  gemm_out_kernel<<<(n + 127) / 128, blk, 0, stream>>>(h, Wout, bout, out, n);
}
